// LackAttention_51625506898453
// MI455X (gfx1250) — compile-verified, matrix operand order checked by reading, not run on hardware
//
#include <hip/hip_runtime.h>
#include <hip/hip_bf16.h>

// ---------------------------------------------------------------------------
// ESIM cross-attention, fused per side, for gfx1250 (MI455X).
//   Pass A: queries = premises   [Np,B,D], keys = hypotheses [Nh,B,D], mask[B,Nh]
//   Pass B: queries = hypotheses [Nh,B,D], keys = premises   [Np,B,D], mask[B,Np]
// Per workgroup: ROWS query rows of one batch. Score strip kept in LDS,
// masked softmax in LDS, weighted sum via a second WMMA GEMM with the key
// matrix staged transposed through LDS in 32-wide K blocks.
// ---------------------------------------------------------------------------

typedef __bf16  bf16;
typedef bf16    v16bf __attribute__((ext_vector_type(16)));
typedef bf16    v8bf  __attribute__((ext_vector_type(8)));
typedef float   v8f   __attribute__((ext_vector_type(8)));

#define BATCH 32
#define DDIM  512
#define NPREM 2048
#define NHYP  1024

// Two contiguous 8-bf16 runs -> one 16-bit A/B WMMA operand (ISA 7.12.2 layout)
__device__ __forceinline__ v16bf ld16(const bf16* p0, const bf16* p1) {
    v8bf a = *(const v8bf*)p0;
    v8bf b = *(const v8bf*)p1;
    return __builtin_shufflevector(a, b, 0,1,2,3,4,5,6,7,8,9,10,11,12,13,14,15);
}

// Build bf16 operand from fp32 row: uses p[0..7] and p[16..23] (runs c0, c0+16)
__device__ __forceinline__ v16bf frag_from_f32(const float* p) {
    float4 a0 = *(const float4*)(p + 0);
    float4 a1 = *(const float4*)(p + 4);
    float4 b0 = *(const float4*)(p + 16);
    float4 b1 = *(const float4*)(p + 20);
    v16bf r;
    r[0]=(bf16)a0.x;  r[1]=(bf16)a0.y;  r[2]=(bf16)a0.z;  r[3]=(bf16)a0.w;
    r[4]=(bf16)a1.x;  r[5]=(bf16)a1.y;  r[6]=(bf16)a1.z;  r[7]=(bf16)a1.w;
    r[8]=(bf16)b0.x;  r[9]=(bf16)b0.y;  r[10]=(bf16)b0.z; r[11]=(bf16)b0.w;
    r[12]=(bf16)b1.x; r[13]=(bf16)b1.y; r[14]=(bf16)b1.z; r[15]=(bf16)b1.w;
    return r;
}

__device__ __forceinline__ float wred_max(float v) {
    #pragma unroll
    for (int o = 16; o > 0; o >>= 1) v = fmaxf(v, __shfl_xor(v, o, 32));
    return v;
}
__device__ __forceinline__ float wred_sum(float v) {
    #pragma unroll
    for (int o = 16; o > 0; o >>= 1) v += __shfl_xor(v, o, 32);
    return v;
}

template <int ROWS, int NKEYS>
__global__ __launch_bounds__(256)
void esim_attend(const float* __restrict__ Q,     // [NQ,    BATCH, DDIM]
                 const float* __restrict__ Kmat,  // [NKEYS, BATCH, DDIM]
                 const int*   __restrict__ mask,  // [BATCH, NKEYS]
                 float*       __restrict__ out,   // [BATCH, NQ, DDIM]
                 int NQ) {
    constexpr int ROWTILES = ROWS / 16;
    const int tid  = threadIdx.x;
    const int lane = tid & 31;
    const int wave = tid >> 5;
    const int qb   = blockIdx.x;       // query row-block
    const int b    = blockIdx.y;       // batch

    extern __shared__ char smem[];
    bf16*  sA  = (bf16*)smem;                                          // ROWS  x DDIM   bf16
    float* sS  = (float*)(smem + (size_t)ROWS*DDIM*2);                 // ROWS  x NKEYS  f32
    bf16*  sW  = (bf16*)((char*)sS + (size_t)ROWS*NKEYS*4);            // ROWS  x NKEYS  bf16
    bf16*  sBt = (bf16*)((char*)sW + (size_t)ROWS*NKEYS*2);            // DDIM  x 32     bf16 (K-block, transposed)

    // ---- stage query rows -> bf16 in LDS ----------------------------------
    for (int lin = tid; lin < ROWS * (DDIM/4); lin += 256) {
        int r  = lin >> 7;              // DDIM/4 = 128
        int d4 = lin & 127;
        const float* src = Q + ((size_t)(qb*ROWS + r)*BATCH + b)*DDIM + d4*4;
        float4 v = *(const float4*)src;
        bf16* dst = sA + r*DDIM + d4*4;
        dst[0]=(bf16)v.x; dst[1]=(bf16)v.y; dst[2]=(bf16)v.z; dst[3]=(bf16)v.w;
    }
    __syncthreads();

    const int c0 = (lane >> 4) * 8;     // K-run offset per WMMA operand layout
    const int ml = lane & 15;

    // ---- phase 1: S = Q . K^T  (ROWS x NKEYS) -----------------------------
    for (int jt = wave; jt < NKEYS/16; jt += 8) {
        v8f acc[ROWTILES] = {};
        const int krow = jt*16 + ml;
        const float* brow = Kmat + ((size_t)krow*BATCH + b)*DDIM;
        #pragma unroll 4
        for (int kb = 0; kb < DDIM; kb += 32) {
            v16bf bf = frag_from_f32(brow + kb + c0);
            #pragma unroll
            for (int rt = 0; rt < ROWTILES; ++rt) {
                const bf16* ap = sA + (rt*16 + ml)*DDIM + kb + c0;
                v16bf af = ld16(ap, ap + 16);
                acc[rt] = __builtin_amdgcn_wmma_f32_16x16x32_bf16(
                    false, af, false, bf, (short)0, acc[rt], false, false);
            }
        }
        #pragma unroll
        for (int rt = 0; rt < ROWTILES; ++rt) {
            int m0 = rt*16 + 8*(lane >> 4);
            int nn = jt*16 + ml;
            #pragma unroll
            for (int r = 0; r < 8; ++r)
                sS[(size_t)(m0 + r)*NKEYS + nn] = acc[rt][r];
        }
    }
    __syncthreads();

    // ---- phase 2: ESIM masked softmax over keys ---------------------------
    // ref: p = softmax(t*m)*m; p /= sum(p)+eps  =>  w_j = m_j*exp(t_j - M),
    // M = max_j(m_j ? t_j : 0), row = w / (sum_w + eps)
    const int* mrow_base = mask + (size_t)b*NKEYS;
    constexpr int RPW = ROWS / 8;
    for (int rr = 0; rr < RPW; ++rr) {
        int row = wave*RPW + rr;
        const float* srow = sS + (size_t)row*NKEYS;
        float mx = -3.4e38f;
        for (int j = lane; j < NKEYS; j += 32) {
            float v = mrow_base[j] ? srow[j] : 0.0f;
            mx = fmaxf(mx, v);
        }
        mx = wred_max(mx);
        float sum = 0.0f;
        for (int j = lane; j < NKEYS; j += 32)
            if (mrow_base[j]) sum += __expf(srow[j] - mx);
        sum = wred_sum(sum);
        float inv = 1.0f / (sum + 1e-13f);
        bf16* wrow = sW + (size_t)row*NKEYS;
        for (int j = lane; j < NKEYS; j += 32)
            wrow[j] = (bf16)(mrow_base[j] ? __expf(srow[j] - mx) * inv : 0.0f);
    }

    // ---- phase 3: out = W . Kmat  (ROWS x DDIM, K = NKEYS) ----------------
    v8f acc[4][ROWTILES] = {};
    for (int kb = 0; kb < NKEYS; kb += 32) {
        __syncthreads();   // sW ready (first iter) / sBt consumed (later iters)
        // stage 32 key rows transposed: sBt[d][k] = Kmat[kb+k][d] as bf16
        for (int lin = tid; lin < 32 * (DDIM/4); lin += 256) {
            int k  = lin >> 7;
            int d4 = lin & 127;
            const float* src = Kmat + ((size_t)(kb + k)*BATCH + b)*DDIM + d4*4;
            float4 v = *(const float4*)src;
            int d = d4*4;
            sBt[(d+0)*32 + k] = (bf16)v.x;
            sBt[(d+1)*32 + k] = (bf16)v.y;
            sBt[(d+2)*32 + k] = (bf16)v.z;
            sBt[(d+3)*32 + k] = (bf16)v.w;
        }
        __syncthreads();

        v16bf af[ROWTILES];
        #pragma unroll
        for (int rt = 0; rt < ROWTILES; ++rt) {
            const bf16* ap = sW + (size_t)(rt*16 + ml)*NKEYS + kb + c0;
            af[rt] = ld16(ap, ap + 16);
        }
        #pragma unroll
        for (int i = 0; i < 4; ++i) {
            int nt = wave + 8*i;                       // d-column tile
            const bf16* bp = sBt + (nt*16 + ml)*32 + c0;
            v16bf bf = ld16(bp, bp + 16);
            #pragma unroll
            for (int rt = 0; rt < ROWTILES; ++rt)
                acc[i][rt] = __builtin_amdgcn_wmma_f32_16x16x32_bf16(
                    false, af[rt], false, bf, (short)0, acc[i][rt], false, false);
        }
    }

    // ---- write out [B, NQ, DDIM] ------------------------------------------
    #pragma unroll
    for (int i = 0; i < 4; ++i) {
        int d0 = (wave + 8*i)*16 + ml;
        #pragma unroll
        for (int rt = 0; rt < ROWTILES; ++rt) {
            int m0 = rt*16 + 8*(lane >> 4);
            #pragma unroll
            for (int r = 0; r < 8; ++r) {
                int qrow = qb*ROWS + m0 + r;
                out[((size_t)b*NQ + qrow)*DDIM + d0] = acc[i][rt][r];
            }
        }
    }
}

extern "C" void kernel_launch(void* const* d_in, const int* in_sizes, int n_in,
                              void* d_out, int out_size, void* d_ws, size_t ws_size,
                              hipStream_t stream) {
    (void)in_sizes; (void)n_in; (void)out_size; (void)d_ws; (void)ws_size;
    const float* premises        = (const float*)d_in[0];   // [Np, B, D]
    const int*   premises_mask   = (const int*)  d_in[1];   // [B, Nh]
    const float* hypotheses      = (const float*)d_in[2];   // [Nh, B, D]
    const int*   hypotheses_mask = (const int*)  d_in[3];   // [B, Np]

    float* new_premises   = (float*)d_out;                               // [B, Np, D]
    float* new_hypotheses = new_premises + (size_t)BATCH*NPREM*DDIM;     // [B, Nh, D]

    // Pass A: ROWS=32 query rows (premises), NKEYS=1024 (hypotheses). LDS 256KB.
    size_t smemA = (size_t)32*DDIM*2 + (size_t)32*NHYP*4 + (size_t)32*NHYP*2 + (size_t)DDIM*32*2;
    esim_attend<32, NHYP><<<dim3(NPREM/32, BATCH), 256, smemA, stream>>>(
        premises, hypotheses, premises_mask, new_premises, NPREM);

    // Pass B: ROWS=16 query rows (hypotheses), NKEYS=2048 (premises). LDS 240KB.
    size_t smemB = (size_t)16*DDIM*2 + (size_t)16*NPREM*4 + (size_t)16*NPREM*2 + (size_t)DDIM*32*2;
    esim_attend<16, NPREM><<<dim3(NHYP/16, BATCH), 256, smemB, stream>>>(
        hypotheses, premises, hypotheses_mask, new_hypotheses, NHYP);
}
